// MambaBlock_16844861735690
// MI455X (gfx1250) — matrix operand order repaired and checked by reading, not compiled, not run on hardware
//
#include <hip/hip_runtime.h>
#include <hip/hip_bf16.h>

#define D_MODEL 1024
#define D_STATE 16
#define D_CONV  4
#define D_INNER 2048
#define BATCH   2
#define SEQ     1024
#define NTOK    (BATCH*SEQ)     // 2048 tokens
#define XZW     (2*D_INNER)     // 4096 (xz row width)
#define NDBL    (3*D_STATE)     // 48  (x_dbl row width)
#define EPS_RMS 1e-8f

typedef __attribute__((ext_vector_type(16))) _Float16 v16h;
typedef __attribute__((ext_vector_type(8)))  _Float16 v8h;
typedef __attribute__((ext_vector_type(8)))  float    v8f;

__device__ __forceinline__ float siluf(float x) { return x / (1.f + __expf(-x)); }
__device__ __forceinline__ float softplusf(float x) {
  return (x > 20.f) ? x : log1pf(__expf(x));
}

// ---------------------------------------------------------------------------
// f32 -> f16 conversion pass (8 elements / thread). n must be a multiple of 8.
// ---------------------------------------------------------------------------
__global__ __launch_bounds__(256)
void cvt_f16(const float* __restrict__ src, _Float16* __restrict__ dst, int n) {
  int i = (blockIdx.x * blockDim.x + threadIdx.x) * 8;
  if (i >= n) return;
  float4 f0 = *(const float4*)(src + i);
  float4 f1 = *(const float4*)(src + i + 4);
  v8h h;
  h[0] = (_Float16)f0.x; h[1] = (_Float16)f0.y;
  h[2] = (_Float16)f0.z; h[3] = (_Float16)f0.w;
  h[4] = (_Float16)f1.x; h[5] = (_Float16)f1.y;
  h[6] = (_Float16)f1.z; h[7] = (_Float16)f1.w;
  *(v8h*)(dst + i) = h;
}

// ---------------------------------------------------------------------------
// A-fragment: 16x32 f16 tile from f16 row-major [M][K].
// lane = half*16 + m ; element e holds
//   K = k0 + 8*half + e           (e = 0..7)
//   K = k0 + 16 + 8*half + (e-8)  (e = 8..15)
// (CDNA5 ISA 7.12.2, 16-bit A-matrix 16x32 table) — 2 x b128 loads.
// ---------------------------------------------------------------------------
__device__ __forceinline__ v16h load_a_frag(const _Float16* __restrict__ P, int ld,
                                            int row, int k0, int half) {
  const _Float16* p = P + (size_t)row * ld + k0 + 8 * half;
  v8h lo = *(const v8h*)(p);
  v8h hi = *(const v8h*)(p + 16);
  v16h a;
  #pragma unroll
  for (int i = 0; i < 8; ++i) { a[i] = lo[i]; a[8 + i] = hi[i]; }
  return a;
}

// ---------------------------------------------------------------------------
// B-fragment: 32x16 f16 tile; source is weight W[N][K] row-major (B = W^T).
// lane = half*16 + n ; element e holds K = k0 + 8*half + e (e<8),
// K = k0 + 16 + 8*half + (e-8) (e>=8) -> 2 x b128 loads.
// ---------------------------------------------------------------------------
__device__ __forceinline__ v16h load_b_frag(const _Float16* __restrict__ P, int ld,
                                            int row, int k0, int half) {
  const _Float16* p = P + (size_t)row * ld + k0 + 8 * half;
  v8h lo = *(const v8h*)(p);
  v8h hi = *(const v8h*)(p + 16);
  v16h b;
  #pragma unroll
  for (int i = 0; i < 8; ++i) { b[i] = lo[i]; b[8 + i] = hi[i]; }
  return b;
}

// ---------------------------------------------------------------------------
// C[M][N] = A[M][K] * W[N][K]^T  — f16 WMMA, f32 accumulate.
// Block = 128 threads = 4 waves arranged 2x2; each wave owns a 32x64 tile
// (2 A-frags x 4 B-frags = 8 WMMAs, 12 b128 loads per k-step). Out-of-range
// rows/cols are clamped to row 0 for loads (garbage only reaches guarded
// stores), keeping EXEC all-1s through every WMMA.
// ---------------------------------------------------------------------------
__global__ __launch_bounds__(128)
void gemm_f16_wmma(const _Float16* __restrict__ A, const _Float16* __restrict__ W,
                   float* __restrict__ C, int M, int N, int K) {
  const int wave = threadIdx.x >> 5;
  const int lane = threadIdx.x & 31;
  const int half = lane >> 4;
  const int lidx = lane & 15;
  const int m0 = blockIdx.y * 64 + (wave >> 1) * 32;
  const int n0 = blockIdx.x * 128 + (wave & 1) * 64;

  const int rA0 = m0 + lidx,      rA1 = rA0 + 16;
  const int cB0 = n0 + lidx,      cB1 = cB0 + 16;
  const int cB2 = cB0 + 32,       cB3 = cB0 + 48;
  const int rsA0 = (rA0 < M) ? rA0 : 0;
  const int rsA1 = (rA1 < M) ? rA1 : 0;
  const int csB0 = (cB0 < N) ? cB0 : 0;
  const int csB1 = (cB1 < N) ? cB1 : 0;
  const int csB2 = (cB2 < N) ? cB2 : 0;
  const int csB3 = (cB3 < N) ? cB3 : 0;

  v8f acc00 = {0,0,0,0,0,0,0,0}, acc01 = {0,0,0,0,0,0,0,0};
  v8f acc02 = {0,0,0,0,0,0,0,0}, acc03 = {0,0,0,0,0,0,0,0};
  v8f acc10 = {0,0,0,0,0,0,0,0}, acc11 = {0,0,0,0,0,0,0,0};
  v8f acc12 = {0,0,0,0,0,0,0,0}, acc13 = {0,0,0,0,0,0,0,0};

  for (int k0 = 0; k0 < K; k0 += 32) {
    v16h a0 = load_a_frag(A, K, rsA0, k0, half);
    v16h a1 = load_a_frag(A, K, rsA1, k0, half);
    v16h b0 = load_b_frag(W, K, csB0, k0, half);
    v16h b1 = load_b_frag(W, K, csB1, k0, half);
    v16h b2 = load_b_frag(W, K, csB2, k0, half);
    v16h b3 = load_b_frag(W, K, csB3, k0, half);
    acc00 = __builtin_amdgcn_wmma_f32_16x16x32_f16(false, a0, false, b0, (short)0, acc00, false, false);
    acc01 = __builtin_amdgcn_wmma_f32_16x16x32_f16(false, a0, false, b1, (short)0, acc01, false, false);
    acc02 = __builtin_amdgcn_wmma_f32_16x16x32_f16(false, a0, false, b2, (short)0, acc02, false, false);
    acc03 = __builtin_amdgcn_wmma_f32_16x16x32_f16(false, a0, false, b3, (short)0, acc03, false, false);
    acc10 = __builtin_amdgcn_wmma_f32_16x16x32_f16(false, a1, false, b0, (short)0, acc10, false, false);
    acc11 = __builtin_amdgcn_wmma_f32_16x16x32_f16(false, a1, false, b1, (short)0, acc11, false, false);
    acc12 = __builtin_amdgcn_wmma_f32_16x16x32_f16(false, a1, false, b2, (short)0, acc12, false, false);
    acc13 = __builtin_amdgcn_wmma_f32_16x16x32_f16(false, a1, false, b3, (short)0, acc13, false, false);
  }

  // C/D layout: VGPR r, lane = half*16 + n -> (M = r + 8*half, N = n)
  const int nc = n0 + lidx;
  #pragma unroll
  for (int r = 0; r < 8; ++r) {
    int mA = m0 + r + 8 * half;
    int mB = mA + 16;
    if (mA < M) {
      if (nc      < N) C[(size_t)mA * N + nc]      = acc00[r];
      if (nc + 16 < N) C[(size_t)mA * N + nc + 16] = acc01[r];
      if (nc + 32 < N) C[(size_t)mA * N + nc + 32] = acc02[r];
      if (nc + 48 < N) C[(size_t)mA * N + nc + 48] = acc03[r];
    }
    if (mB < M) {
      if (nc      < N) C[(size_t)mB * N + nc]      = acc10[r];
      if (nc + 16 < N) C[(size_t)mB * N + nc + 16] = acc11[r];
      if (nc + 32 < N) C[(size_t)mB * N + nc + 32] = acc12[r];
      if (nc + 48 < N) C[(size_t)mB * N + nc + 48] = acc13[r];
    }
  }
}

// ---------------------------------------------------------------------------
// Depthwise causal conv (4 taps) + bias + SiLU; writes f32 (for the scan)
// and f16 (for the x_proj WMMA GEMM).
// ---------------------------------------------------------------------------
__global__ __launch_bounds__(256)
void conv_silu(const float* __restrict__ xz, const float* __restrict__ cw,
               const float* __restrict__ cb, float* __restrict__ xc,
               _Float16* __restrict__ xc16) {
  int idx = blockIdx.x * blockDim.x + threadIdx.x;
  if (idx >= NTOK * D_INNER) return;
  int c = idx % D_INNER;
  int t = idx / D_INNER;    // token = b*SEQ + l
  int l = t % SEQ;
  float acc = cb[c];
  #pragma unroll
  for (int j = 0; j < D_CONV; ++j) {
    int ll = l - (D_CONV - 1) + j;
    if (ll >= 0)
      acc += cw[c * D_CONV + j] * xz[(size_t)(t - (D_CONV - 1) + j) * XZW + c];
  }
  float v = siluf(acc);
  xc[(size_t)t * D_INNER + c]   = v;
  xc16[(size_t)t * D_INNER + c] = (_Float16)v;
}

// ---------------------------------------------------------------------------
// Selective scan, one thread per (batch, channel d). dt_proj + softplus fused.
// x_dbl row = [dlt(16) | B(16) | C(16)].
// ---------------------------------------------------------------------------
__global__ __launch_bounds__(256)
void scan_kernel(const float* __restrict__ xc, const float* __restrict__ xdbl,
                 const float* __restrict__ dtw, const float* __restrict__ dtb,
                 const float* __restrict__ Alog, const float* __restrict__ Dp,
                 float* __restrict__ y) {
  int d = blockIdx.x * blockDim.x + threadIdx.x;
  int b = blockIdx.y;
  if (d >= D_INNER) return;

  float Arow[D_STATE], dw[D_STATE], h[D_STATE];
  #pragma unroll
  for (int n = 0; n < D_STATE; ++n) {
    Arow[n] = -__expf(Alog[d * D_STATE + n]);
    dw[n]   = dtw[d * D_STATE + n];
    h[n]    = 0.f;
  }
  const float db = dtb[d];
  const float Dd = Dp[d];

  for (int l = 0; l < SEQ; ++l) {
    int t = b * SEQ + l;
    const float* __restrict__ row = xdbl + (size_t)t * NDBL;
    float u = xc[(size_t)t * D_INNER + d];
    float dpre = db;
    #pragma unroll
    for (int n = 0; n < D_STATE; ++n) dpre += dw[n] * row[n];
    float dt = softplusf(dpre);
    float yv = 0.f;
    #pragma unroll
    for (int n = 0; n < D_STATE; ++n) {
      float dA = __expf(dt * Arow[n]);
      h[n] = dA * h[n] + (dt * row[D_STATE + n]) * u;
      yv  += h[n] * row[2 * D_STATE + n];
    }
    y[(size_t)t * D_INNER + d] = yv + u * Dd;
  }
}

// ---------------------------------------------------------------------------
// y16 = f16( RMSNorm(y * silu(z)) * norm_w ). One block (256 thr) per token.
// ---------------------------------------------------------------------------
__global__ __launch_bounds__(256)
void gated_norm(const float* __restrict__ xz, const float* __restrict__ nw,
                const float* __restrict__ y, _Float16* __restrict__ y16) {
  __shared__ float sdata[256];
  const int t = blockIdx.x;
  float vals[D_INNER / 256];
  float ss = 0.f;
  #pragma unroll
  for (int i = 0; i < D_INNER / 256; ++i) {
    int c = threadIdx.x + i * 256;
    float z = xz[(size_t)t * XZW + D_INNER + c];
    float v = y[(size_t)t * D_INNER + c] * siluf(z);
    vals[i] = v;
    ss += v * v;
  }
  sdata[threadIdx.x] = ss;
  __syncthreads();
  for (int s = 128; s > 0; s >>= 1) {
    if (threadIdx.x < s) sdata[threadIdx.x] += sdata[threadIdx.x + s];
    __syncthreads();
  }
  float scale = rsqrtf(sdata[0] / (float)D_INNER + EPS_RMS);
  #pragma unroll
  for (int i = 0; i < D_INNER / 256; ++i) {
    int c = threadIdx.x + i * 256;
    y16[(size_t)t * D_INNER + c] = (_Float16)(vals[i] * scale * nw[c]);
  }
}

// ---------------------------------------------------------------------------
extern "C" void kernel_launch(void* const* d_in, const int* in_sizes, int n_in,
                              void* d_out, int out_size, void* d_ws, size_t ws_size,
                              hipStream_t stream) {
  (void)in_sizes; (void)n_in; (void)out_size; (void)ws_size;

  const float* x       = (const float*)d_in[0];   // [B,L,1024]
  const float* in_w    = (const float*)d_in[1];   // [4096,1024]
  const float* conv_w  = (const float*)d_in[2];   // [2048,1,4]
  const float* conv_b  = (const float*)d_in[3];   // [2048]
  const float* xproj_w = (const float*)d_in[4];   // [48,2048]
  const float* dt_w    = (const float*)d_in[5];   // [2048,16]
  const float* dt_b    = (const float*)d_in[6];   // [2048]
  const float* A_log   = (const float*)d_in[7];   // [2048,16]
  const float* Dp      = (const float*)d_in[8];   // [2048]
  const float* norm_w  = (const float*)d_in[9];   // [2048]
  const float* out_w   = (const float*)d_in[10];  // [1024,2048]
  float* out = (float*)d_out;                     // [B,L,1024]

  // ---- workspace layout (f32 region, then 16B-aligned f16 region) ----
  float* ws   = (float*)d_ws;
  float* xz   = ws;                                   // NTOK*4096 f32
  float* xc   = xz   + (size_t)NTOK * XZW;            // NTOK*2048 f32
  float* xdbl = xc   + (size_t)NTOK * D_INNER;        // NTOK*48   f32
  float* y    = xdbl + (size_t)NTOK * NDBL;           // NTOK*2048 f32
  _Float16* x16   = (_Float16*)(y + (size_t)NTOK * D_INNER);
  _Float16* wi16  = x16  + (size_t)NTOK * D_MODEL;    // 4096*1024
  _Float16* wxp16 = wi16 + (size_t)XZW * D_MODEL;     // 48*2048
  _Float16* wo16  = wxp16 + (size_t)NDBL * D_INNER;   // 1024*2048
  _Float16* xc16  = wo16 + (size_t)D_MODEL * D_INNER; // NTOK*2048
  _Float16* y16   = xc16 + (size_t)NTOK * D_INNER;    // NTOK*2048

  // 0) one-time (per call) f32 -> f16 conversions
  const int nX  = NTOK * D_MODEL;      // 2M
  const int nWi = XZW * D_MODEL;       // 4M
  const int nWx = NDBL * D_INNER;      // 96K
  const int nWo = D_MODEL * D_INNER;   // 2M
  cvt_f16<<<(nX  / 8 + 255) / 256, 256, 0, stream>>>(x,       x16,   nX);
  cvt_f16<<<(nWi / 8 + 255) / 256, 256, 0, stream>>>(in_w,    wi16,  nWi);
  cvt_f16<<<(nWx / 8 + 255) / 256, 256, 0, stream>>>(xproj_w, wxp16, nWx);
  cvt_f16<<<(nWo / 8 + 255) / 256, 256, 0, stream>>>(out_w,   wo16,  nWo);

  // 1) xz = x @ in_proj_w.T              [2048 x 4096], K=1024
  gemm_f16_wmma<<<dim3(XZW / 128, NTOK / 64), 128, 0, stream>>>(
      x16, wi16, xz, NTOK, XZW, D_MODEL);

  // 2) xc = silu(causal_dwconv(xz[:, :2048]) + b)   (f32 + f16 copies)
  conv_silu<<<(NTOK * D_INNER) / 256, 256, 0, stream>>>(xz, conv_w, conv_b, xc, xc16);

  // 3) x_dbl = xc @ x_proj_w.T           [2048 x 48], K=2048 (N guarded)
  gemm_f16_wmma<<<dim3(1, NTOK / 64), 128, 0, stream>>>(
      xc16, wxp16, xdbl, NTOK, NDBL, D_INNER);

  // 4) selective scan with fused dt_proj + softplus
  scan_kernel<<<dim3(D_INNER / 256, BATCH), 256, 0, stream>>>(
      xc, xdbl, dt_w, dt_b, A_log, Dp, y);

  // 5) y16 = f16( RMSNorm(y * silu(z)) * norm_w )
  gated_norm<<<NTOK, 256, 0, stream>>>(xz, norm_w, y, y16);

  // 6) out = y @ out_proj_w.T            [2048 x 1024], K=2048
  gemm_f16_wmma<<<dim3(D_MODEL / 128, NTOK / 64), 128, 0, stream>>>(
      y16, wo16, out, NTOK, D_MODEL, D_INNER);
}
